// attention_node_to_node_12034498363711
// MI455X (gfx1250) — matrix operand order repaired and checked by reading, not compiled, not run on hardware
//
#include <hip/hip_runtime.h>

typedef __attribute__((ext_vector_type(16))) _Float16 v16h;
typedef __attribute__((ext_vector_type(8)))  float    v8f;

#define BB 8
#define NN 512
#define DD 256
#define DH 64
#define NEG_INF_F (-1.0e9f)

// A-matrix (16x32 f16) element index for lane-half h, packed half i: contiguous K = 16h+i
__device__ __forceinline__ int klocal(int h, int i) {
  return 16 * h + i;
}

// ---------------------------------------------------------------------------
// Kernel 1: Q/K/V projections. One wave per 16-row tile of X (B*N=4096 rows).
//   Q,K stored (B*N, DH) f16; V stored transposed (B, DH, N) f16.
// ---------------------------------------------------------------------------
__global__ __launch_bounds__(32)
void qkv_wmma_kernel(const float* __restrict__ x, const int* __restrict__ nm,
                     const float* __restrict__ Wq, const float* __restrict__ bq,
                     const float* __restrict__ Wk, const float* __restrict__ bk,
                     const float* __restrict__ Wv, const float* __restrict__ bv,
                     _Float16* __restrict__ Qh, _Float16* __restrict__ Kh,
                     _Float16* __restrict__ Vt) {
  const int lane = threadIdx.x;
  const int nl = lane & 15, h = lane >> 4;
  const int row0 = blockIdx.x * 16;

  // Preload all 8 K-chunks of the A tile (row m = nl)
  v16h A[8];
  const float* xr = x + (size_t)(row0 + nl) * DD;
#pragma unroll
  for (int kc = 0; kc < 8; ++kc) {
#pragma unroll
    for (int i = 0; i < 16; ++i)
      A[kc][i] = (_Float16)xr[kc * 32 + klocal(h, i)];
  }

  const float* Ws[3] = {Wq, Wk, Wv};
  const float* bs[3] = {bq, bk, bv};

  for (int mat = 0; mat < 3; ++mat) {
    const float* W    = Ws[mat];
    const float* bias = bs[mat];
#pragma unroll
    for (int nt = 0; nt < 4; ++nt) {
      const int col = nt * 16 + nl;
      v8f acc = {};
#pragma unroll
      for (int kc = 0; kc < 8; ++kc) {
        v16h Bt;  // B-layout: lane holds column `col`, halves i -> K = 16h+i
#pragma unroll
        for (int i = 0; i < 16; ++i)
          Bt[i] = (_Float16)W[(size_t)(kc * 32 + 16 * h + i) * DH + col];
        acc = __builtin_amdgcn_wmma_f32_16x16x32_f16(false, A[kc], false, Bt,
                                                     (short)0, acc, false, false);
      }
      const float bcol = bias[col];
#pragma unroll
      for (int r = 0; r < 8; ++r) {           // D: row m = r + 8h, col = nl-based
        const int grow = row0 + r + 8 * h;
        const float mk  = (float)nm[grow];
        const float val = (acc[r] + bcol) * mk;
        if (mat == 0) {
          Qh[(size_t)grow * DH + col] = (_Float16)val;
        } else if (mat == 1) {
          Kh[(size_t)grow * DH + col] = (_Float16)val;
        } else {
          const int bidx = grow >> 9, j = grow & (NN - 1);
          Vt[((size_t)bidx * DH + col) * NN + j] = (_Float16)val;  // transposed
        }
      }
    }
  }
}

// ---------------------------------------------------------------------------
// Kernel 2: flash attention. One wave per (batch, 16-row tile).
// K/V chunk tiles staged Global->LDS with async-LDS DMA (ASYNCcnt path).
// ---------------------------------------------------------------------------
#define ASYNC_B128(LDS32, GA64, OFF)                                           \
  asm volatile("global_load_async_to_lds_b128 %0, %1, off offset:" #OFF        \
               :: "v"(LDS32), "v"(GA64) : "memory")

__global__ __launch_bounds__(32)
void attn_wmma_kernel(const _Float16* __restrict__ Qh, const _Float16* __restrict__ Kh,
                      const _Float16* __restrict__ Vt, const int* __restrict__ nm,
                      float* __restrict__ out) {
  __shared__ _Float16 Pl[16 * 32];   // P chunk staging for D->A relayout
  __shared__ _Float16 Kb[32 * 64];   // K chunk: 32 keys x 64 dh  (4 KB)
  __shared__ _Float16 Vb[64 * 32];   // V chunk: 64 dh x 32 keys  (4 KB, transposed)
  const int lane = threadIdx.x;
  const int nl = lane & 15, h = lane >> 4;
  const int b  = blockIdx.x >> 5;          // 32 row-tiles per batch
  const int m0 = (blockIdx.x & 31) * 16;

  // Raw LDS byte offsets (flat LDS address low 32 bits == wave LDS offset)
  const unsigned ldsK = (unsigned)(size_t)(void*)Kb;
  const unsigned ldsV = (unsigned)(size_t)(void*)Vb;

  // Q tile as two A chunks (dh = 64 = 2 x K32)
  v16h AQ[2];
  const _Float16* qr = Qh + (size_t)(b * NN + m0 + nl) * DH;
#pragma unroll
  for (int kc = 0; kc < 2; ++kc) {
#pragma unroll
    for (int i = 0; i < 16; ++i)
      AQ[kc][i] = qr[kc * 32 + klocal(h, i)];
  }

  v8f O[4];
  v8f zero = {};
#pragma unroll
  for (int nt = 0; nt < 4; ++nt) O[nt] = zero;

  float mrow[8], srow[8];
#pragma unroll
  for (int r = 0; r < 8; ++r) { mrow[r] = -INFINITY; srow[r] = 0.0f; }

  for (int jc = 0; jc < 16; ++jc) {        // 16 chunks of 32 keys
    const int j0 = jc * 32;

    // ---- async-DMA stage: K chunk (lane -> key row `lane`, 128 B)
    //      and V chunk (lane -> dh cols `lane` and `lane+32`, 64 B each) ----
    asm volatile("s_wait_dscnt 0x0" ::: "memory");   // prior LDS reads retired
    {
      const unsigned long long kg =
          (unsigned long long)(size_t)(Kh + (size_t)(b * NN + j0 + lane) * DH);
      const unsigned kdst = ldsK + (unsigned)lane * 128u;
      ASYNC_B128(kdst, kg, 0);   ASYNC_B128(kdst, kg, 16);
      ASYNC_B128(kdst, kg, 32);  ASYNC_B128(kdst, kg, 48);
      ASYNC_B128(kdst, kg, 64);  ASYNC_B128(kdst, kg, 80);
      ASYNC_B128(kdst, kg, 96);  ASYNC_B128(kdst, kg, 112);

      const unsigned long long vg0 =
          (unsigned long long)(size_t)(Vt + ((size_t)(b * DH + lane)) * NN + j0);
      const unsigned long long vg1 =
          (unsigned long long)(size_t)(Vt + ((size_t)(b * DH + lane + 32)) * NN + j0);
      const unsigned vdst0 = ldsV + (unsigned)lane * 64u;
      const unsigned vdst1 = ldsV + (unsigned)(lane + 32) * 64u;
      ASYNC_B128(vdst0, vg0, 0);  ASYNC_B128(vdst0, vg0, 16);
      ASYNC_B128(vdst0, vg0, 32); ASYNC_B128(vdst0, vg0, 48);
      ASYNC_B128(vdst1, vg1, 0);  ASYNC_B128(vdst1, vg1, 16);
      ASYNC_B128(vdst1, vg1, 32); ASYNC_B128(vdst1, vg1, 48);
    }
    asm volatile("s_wait_asynccnt 0x0" ::: "memory");  // tiles landed in LDS

    // ---- S = Q K^T for this chunk (from LDS) ----
    v8f S[2];
    int mj[2];
#pragma unroll
    for (int jn = 0; jn < 2; ++jn) {
      const int jl = 16 * jn + nl;               // chunk-local key for this column
      mj[jn] = nm[b * NN + j0 + jl];
      const _Float16* kr = Kb + jl * 64;         // B-layout: contiguous halves
      v16h BK0, BK1;
#pragma unroll
      for (int i = 0; i < 16; ++i) {
        BK0[i] = kr[16 * h + i];
        BK1[i] = kr[32 + 16 * h + i];
      }
      v8f sacc = {};
      sacc = __builtin_amdgcn_wmma_f32_16x16x32_f16(false, AQ[0], false, BK0,
                                                    (short)0, sacc, false, false);
      sacc = __builtin_amdgcn_wmma_f32_16x16x32_f16(false, AQ[1], false, BK1,
                                                    (short)0, sacc, false, false);
      S[jn] = sacc;
    }

    // ---- masked online softmax ----
    float corr[8];
#pragma unroll
    for (int r = 0; r < 8; ++r) {
      float v0 = mj[0] ? S[0][r] * 0.125f : NEG_INF_F;  // 1/sqrt(64)
      float v1 = mj[1] ? S[1][r] * 0.125f : NEG_INF_F;
      float cm = fmaxf(v0, v1);
#pragma unroll
      for (int off = 1; off < 16; off <<= 1)
        cm = fmaxf(cm, __shfl_xor(cm, off, 32));        // reduce over 16-lane half
      const float mn = fmaxf(mrow[r], cm);
      corr[r] = __expf(mrow[r] - mn);                   // exp(-inf - finite) = 0
      const float p0 = __expf(v0 - mn);
      const float p1 = __expf(v1 - mn);
      float ps = p0 + p1;
#pragma unroll
      for (int off = 1; off < 16; off <<= 1)
        ps += __shfl_xor(ps, off, 32);
      srow[r] = srow[r] * corr[r] + ps;
      mrow[r] = mn;
      Pl[(r + 8 * h) * 32 + nl]      = (_Float16)p0;    // D-layout -> LDS
      Pl[(r + 8 * h) * 32 + 16 + nl] = (_Float16)p1;
    }
    __syncthreads();

    v16h AP;                                            // reload P in A-layout
#pragma unroll
    for (int i = 0; i < 16; ++i)
      AP[i] = Pl[nl * 32 + klocal(h, i)];

    // ---- O = corr*O + P V (V chunk from LDS) ----
#pragma unroll
    for (int nt = 0; nt < 4; ++nt) {
      const _Float16* vr = Vb + (nt * 16 + nl) * 32 + 16 * h;
      v16h BV;
#pragma unroll
      for (int i = 0; i < 16; ++i) BV[i] = vr[i];
      v8f Oc;
#pragma unroll
      for (int r = 0; r < 8; ++r) Oc[r] = O[nt][r] * corr[r];
      O[nt] = __builtin_amdgcn_wmma_f32_16x16x32_f16(false, AP, false, BV,
                                                     (short)0, Oc, false, false);
    }
    __syncthreads();
  }

#pragma unroll
  for (int nt = 0; nt < 4; ++nt) {
#pragma unroll
    for (int r = 0; r < 8; ++r) {
      const int grow = b * NN + m0 + r + 8 * h;
      const float mk = (float)nm[grow];                 // out *= x_mask
      out[(size_t)grow * DH + nt * 16 + nl] = O[nt][r] / srow[r] * mk;
    }
  }
}

// ---------------------------------------------------------------------------
// Kernel 3: passthrough copy of `e` (HBM-roofline bound: 537 MB of traffic).
// ---------------------------------------------------------------------------
__global__ __launch_bounds__(256)
void copy_e_kernel(const float4* __restrict__ src, float4* __restrict__ dst, int n4) {
  size_t i = (size_t)blockIdx.x * blockDim.x + threadIdx.x;
  const size_t stride = (size_t)gridDim.x * blockDim.x;
  for (; i < (size_t)n4; i += stride) {
    __builtin_prefetch((const void*)(src + i + stride), 0, 1);  // global_prefetch_b8
    dst[i] = src[i];
  }
}

// ---------------------------------------------------------------------------
extern "C" void kernel_launch(void* const* d_in, const int* in_sizes, int n_in,
                              void* d_out, int out_size, void* d_ws, size_t ws_size,
                              hipStream_t stream) {
  (void)in_sizes; (void)n_in; (void)out_size; (void)ws_size;
  const float* x  = (const float*)d_in[0];
  const float* e  = (const float*)d_in[1];
  const int*   nm = (const int*)d_in[2];
  const float* Wq = (const float*)d_in[3];
  const float* bq = (const float*)d_in[4];
  const float* Wk = (const float*)d_in[5];
  const float* bk = (const float*)d_in[6];
  const float* Wv = (const float*)d_in[7];
  const float* bv = (const float*)d_in[8];
  float* out = (float*)d_out;

  _Float16* Qh = (_Float16*)d_ws;                 // 512 KB
  _Float16* Kh = Qh + (size_t)BB * NN * DH;       // 512 KB
  _Float16* Vt = Kh + (size_t)BB * NN * DH;       // 512 KB (transposed V)

  qkv_wmma_kernel<<<(BB * NN) / 16, 32, 0, stream>>>(x, nm, Wq, bq, Wk, bk, Wv, bv,
                                                     Qh, Kh, Vt);
  attn_wmma_kernel<<<BB * (NN / 16), 32, 0, stream>>>(Qh, Kh, Vt, nm, out);

  const int n4 = (BB * NN * NN * 32) / 4;         // 16,777,216 float4s
  copy_e_kernel<<<2048, 256, 0, stream>>>((const float4*)e,
                                          (float4*)(out + (size_t)BB * NN * DH), n4);
}
